// graphNN_1529008357863
// MI455X (gfx1250) — matrix operand order repaired and checked by reading, not compiled, not run on hardware
//
#include <hip/hip_runtime.h>
#include <hip/hip_bf16.h>

// ---------------------------------------------------------------------------
// CDNA5 (gfx1250) GNN pipeline: edge-message GEMM + scatter, TAGConv k=2 x2,
// mean-pool + predict.  All GEMMs use v_wmma_f32_16x16x32_f16 (f16 in / f32
// accumulate); all segment sums use f32 global atomics.
// ---------------------------------------------------------------------------

typedef __attribute__((ext_vector_type(16))) _Float16 v16h;
typedef __attribute__((ext_vector_type(8)))  float    v8f;

union V16H { uint4 u[2]; v16h v; };

#define N_NODES 50000
#define N_EDGES 1600000
#define NODE_DIM 128
#define EDGE_DIM 64
#define HIDDEN 128
#define TILES_PER_BLOCK 8   // edge tiles (16 edges each) per block

static inline int cdiv(long long a, long long b) { return (int)((a + b - 1) / b); }

// ---------------------------- small utility kernels ------------------------

__global__ void k_zero_f32(float* __restrict__ p, int count) {
  int i = blockIdx.x * blockDim.x + threadIdx.x;
  if (i < count) p[i] = 0.0f;
}

__global__ void k_f32_to_f16(const float* __restrict__ in, _Float16* __restrict__ out, int count) {
  int i = blockIdx.x * blockDim.x + threadIdx.x;
  if (i < count) out[i] = (_Float16)in[i];
}

// in: [K][Ncol] f32 row-major  ->  out: [Ncol][K] f16  (B ready for WMMA)
__global__ void k_transpose_w(const float* __restrict__ in, _Float16* __restrict__ out,
                              int K, int Ncol) {
  int idx = blockIdx.x * blockDim.x + threadIdx.x;
  if (idx >= K * Ncol) return;
  int k = idx / Ncol;
  int n = idx - k * Ncol;
  out[(size_t)n * K + k] = (_Float16)in[idx];
}

__global__ void k_deg(const int* __restrict__ dst, float* __restrict__ deg, int E) {
  int e = blockIdx.x * blockDim.x + threadIdx.x;
  if (e < E) atomicAdd(&deg[dst[e]], 1.0f);
}

__global__ void k_norm(const float* __restrict__ deg, float* __restrict__ norm, int N) {
  int i = blockIdx.x * blockDim.x + threadIdx.x;
  if (i < N) {
    float d = deg[i];
    d = d < 1.0f ? 1.0f : d;
    norm[i] = rsqrtf(d);
  }
}

// x0 = concat(node_fea, h_neigh): write f16 copy and y = x0 * norm (f32)
__global__ void k_build_x0(const float* __restrict__ node_fea,
                           const float* __restrict__ h_neigh,
                           const float* __restrict__ norm,
                           _Float16* __restrict__ x016,
                           float* __restrict__ y, int N) {
  int idx = blockIdx.x * blockDim.x + threadIdx.x;
  if (idx >= N * 256) return;
  int i = idx >> 8;
  int c = idx & 255;
  float v = (c < 128) ? node_fea[(size_t)i * 128 + c]
                      : h_neigh[(size_t)i * 128 + (c - 128)];
  x016[idx] = (_Float16)v;
  y[idx] = v * norm[i];
}

// y = x * norm  (generic, D = 1<<dshift)
__global__ void k_scale(const float* __restrict__ x, const float* __restrict__ norm,
                        float* __restrict__ y, int N, int dshift) {
  int idx = blockIdx.x * blockDim.x + threadIdx.x;
  if (idx >= (N << dshift)) return;
  int i = idx >> dshift;
  y[idx] = x[idx] * norm[i];
}

// scatter-add: acc[dst[e], :] += y[src[e], :]   (D = 4 << chunkShift floats)
__global__ void k_scatter(const float* __restrict__ y,
                          const int* __restrict__ src, const int* __restrict__ dst,
                          float* __restrict__ acc, int E, int chunkShift) {
  long long idx = (long long)blockIdx.x * blockDim.x + threadIdx.x;
  long long e = idx >> chunkShift;
  if (e >= E) return;
  int c4 = (int)(idx & ((1 << chunkShift) - 1));
  int D = 4 << chunkShift;
  int s = src[e], d = dst[e];
  const float4 v = *reinterpret_cast<const float4*>(y + (size_t)s * D + 4 * c4);
  float* o = acc + (size_t)d * D + 4 * c4;
  atomicAdd(o + 0, v.x);
  atomicAdd(o + 1, v.y);
  atomicAdd(o + 2, v.z);
  atomicAdd(o + 3, v.w);
}

// h = norm * acc  -> f16 out; optionally ynext = norm * h (for second hop)
__global__ void k_postprop(const float* __restrict__ acc, const float* __restrict__ norm,
                           _Float16* __restrict__ h16, float* __restrict__ ynext,
                           int N, int dshift) {
  int idx = blockIdx.x * blockDim.x + threadIdx.x;
  if (idx >= (N << dshift)) return;
  int i = idx >> dshift;
  float nm = norm[i];
  float v = nm * acc[idx];
  h16[idx] = (_Float16)v;
  if (ynext) ynext[idx] = nm * v;
}

__global__ void k_mean_partial(const float* __restrict__ h, float* __restrict__ hg, int N) {
  int c = threadIdx.x;  // 0..127
  float s = 0.0f;
  for (int i = blockIdx.x; i < N; i += gridDim.x) s += h[(size_t)i * 128 + c];
  atomicAdd(&hg[c], s);
}

__global__ void k_predict(const float* __restrict__ hg, const float* __restrict__ Wp,
                          const float* __restrict__ bp, float* __restrict__ out, int N) {
  __shared__ float red[128];
  int c = threadIdx.x;
  red[c] = (hg[c] / (float)N) * Wp[c];
  __syncthreads();
  for (int s = 64; s > 0; s >>= 1) {
    if (c < s) red[c] += red[c + s];
    __syncthreads();
  }
  if (c == 0) out[0] = red[0] + bp[0];
}

// ------------------------- edge message GEMM + scatter ----------------------
// One block = 8 waves, TILES_PER_BLOCK tiles of 16 edges (128 edges/block).
// Each wave owns a 16-column slice of msg_W^T kept entirely in registers
// (6 x v16h = 48 VGPRs), loaded from L2 exactly once per block.  Per tile the
// 16x192 f16 A panel is staged in LDS (row stride 200 halves -> conflict-free
// ds_load_b128), multiplied with 6 v_wmma_f32_16x16x32_f16 steps, then ReLU
// and f32 atomic scatter-add into h_neigh[dst].
__global__ __launch_bounds__(256) void edge_msg_kernel(
    const _Float16* __restrict__ node16,   // [N][128] f16
    const float* __restrict__ edge_fea,    // [E][64] f32
    const int* __restrict__ src,
    const int* __restrict__ dst,
    const _Float16* __restrict__ wT,       // [128][192] f16 (msg_W^T)
    const float* __restrict__ bias,        // [128]
    float* __restrict__ h_neigh) {         // [N][128] f32 (pre-zeroed)
  __shared__ _Float16 ldsA[16 * 200];
  __shared__ int ldsDst[16];
  const int tid = threadIdx.x;
  const int w  = tid >> 5;   // wave 0..7 -> N slice
  const int l  = tid & 31;
  const int lm = l & 15;     // A: M row; B/C/D: N col within slice
  const int g  = l >> 4;     // lane group
  const int n  = w * 16 + lm;

  // ---- B slice -> registers (once per block) ----
  V16H B[6];
  {
    const _Float16* brow = wT + (size_t)n * 192;
#pragma unroll
    for (int kk = 0; kk < 6; ++kk) {
      B[kk].u[0] = *reinterpret_cast<const uint4*>(brow + 32 * kk + 8 * g);
      B[kk].u[1] = *reinterpret_cast<const uint4*>(brow + 32 * kk + 16 + 8 * g);
    }
  }
  const float bn = bias[n];

  const int le = tid >> 4;   // staging: edge-in-tile
  const int c  = tid & 15;   // staging: chunk
  const int blockE = blockIdx.x * (16 * TILES_PER_BLOCK);

  for (int t = 0; t < TILES_PER_BLOCK; ++t) {
    const int ebase = blockE + t * 16;
    {  // stage A tile: 16 threads per edge
      const int e = ebase + le;
      const int s = src[e];
      // node half: 8 f16 per chunk
      const uint4 nv = *reinterpret_cast<const uint4*>(node16 + (size_t)s * 128 + 8 * c);
      *reinterpret_cast<uint4*>(&ldsA[le * 200 + 8 * c]) = nv;
      // edge half: 4 f32 -> 4 f16 per chunk (cols 128..191)
      const float4 ef = *reinterpret_cast<const float4*>(edge_fea + (size_t)e * 64 + 4 * c);
      _Float16* ap = &ldsA[le * 200 + 128 + 4 * c];
      ap[0] = (_Float16)ef.x; ap[1] = (_Float16)ef.y;
      ap[2] = (_Float16)ef.z; ap[3] = (_Float16)ef.w;
      if (tid < 16) ldsDst[tid] = dst[ebase + tid];
      // stream-prefetch next tile's edge features (the one true HBM stream)
      if (t + 1 < TILES_PER_BLOCK)
        __builtin_prefetch(edge_fea + (size_t)(e + 16) * 64, 0, 0);
    }
    __syncthreads();

    v8f acc = {};
#pragma unroll
    for (int kk = 0; kk < 6; ++kk) {
      const int k0 = kk * 32;
      V16H A;
      // A 16x32 f16 layout: lane m=l%16; halves 0..7 -> K=k0+8g, 8..15 -> K=k0+16+8g
      A.u[0] = *reinterpret_cast<const uint4*>(&ldsA[lm * 200 + k0 + 8 * g]);
      A.u[1] = *reinterpret_cast<const uint4*>(&ldsA[lm * 200 + k0 + 16 + 8 * g]);
      acc = __builtin_amdgcn_wmma_f32_16x16x32_f16(false, A.v, false, B[kk].v,
                                                   (short)0, acc, false, false);
    }

#pragma unroll
    for (int r = 0; r < 8; ++r) {
      float v = acc[r] + bn;
      v = v > 0.0f ? v : 0.0f;          // ReLU
      const int m = r + 8 * g;          // C/D layout: M = r + 8*(lane/16)
      const int d = ldsDst[m];
      atomicAdd(&h_neigh[(size_t)d * 128 + n], v);
    }
    __syncthreads();  // protect LDS A panel before restaging
  }
}

// ---------------------------- node-side GEMM --------------------------------
// out[m, :] = relu( concat(s0,s1,s2)[m, :] @ W + bias ), W given transposed
// [128][Ktot] f16.  One block = 16 rows x 128 cols (8 waves).  segW = 256/128.
__global__ __launch_bounds__(256) void node_gemm_kernel(
    const _Float16* __restrict__ s0,
    const _Float16* __restrict__ s1,
    const _Float16* __restrict__ s2,
    int segW, int segShift, int nk,        // nk = Ktot/32
    const _Float16* __restrict__ wT,       // [128][Ktot]
    const float* __restrict__ bias,        // [128]
    float* __restrict__ out,               // [M][128]
    _Float16* __restrict__ out16) {        // optional [M][128]
  const int tid = threadIdx.x;
  const int rowbase = blockIdx.x * 16;
  const int w  = tid >> 5;
  const int l  = tid & 31;
  const int lm = l & 15;
  const int g  = l >> 4;
  const int Ktot = nk * 32;
  const int n = w * 16 + lm;
  const _Float16* brow = wT + (size_t)n * Ktot;
  const int m = rowbase + lm;

  v8f acc = {};
  for (int kk = 0; kk < nk; ++kk) {
    const int k0 = kk * 32;
    const int seg = k0 >> segShift;
    const int kin = k0 & (segW - 1);
    const _Float16* sp = (seg == 0) ? s0 : ((seg == 1) ? s1 : s2);
    const _Float16* arow = sp + (size_t)m * segW + kin;
    V16H A, B;
    A.u[0] = *reinterpret_cast<const uint4*>(arow + 8 * g);
    A.u[1] = *reinterpret_cast<const uint4*>(arow + 16 + 8 * g);
    B.u[0] = *reinterpret_cast<const uint4*>(brow + k0 + 8 * g);
    B.u[1] = *reinterpret_cast<const uint4*>(brow + k0 + 16 + 8 * g);
    acc = __builtin_amdgcn_wmma_f32_16x16x32_f16(false, A.v, false, B.v,
                                                 (short)0, acc, false, false);
  }

  const float bn = bias[n];
#pragma unroll
  for (int r = 0; r < 8; ++r) {
    float v = acc[r] + bn;
    v = v > 0.0f ? v : 0.0f;
    const int mr = rowbase + r + 8 * g;
    out[(size_t)mr * 128 + n] = v;
    if (out16) out16[(size_t)mr * 128 + n] = (_Float16)v;
  }
}

// ---------------------------------------------------------------------------

extern "C" void kernel_launch(void* const* d_in, const int* in_sizes, int n_in,
                              void* d_out, int out_size, void* d_ws, size_t ws_size,
                              hipStream_t stream) {
  (void)in_sizes; (void)n_in; (void)out_size; (void)ws_size;

  const float* node_fea = (const float*)d_in[0];
  const float* edge_fea = (const float*)d_in[1];
  const int*   src      = (const int*)d_in[2];
  const int*   dst      = (const int*)d_in[3];
  const float* msg_W    = (const float*)d_in[4];
  const float* msg_b    = (const float*)d_in[5];
  const float* W1       = (const float*)d_in[6];
  const float* b1       = (const float*)d_in[7];
  const float* W2       = (const float*)d_in[8];
  const float* b2       = (const float*)d_in[9];
  const float* Wp       = (const float*)d_in[10];
  const float* bp       = (const float*)d_in[11];
  float* out = (float*)d_out;

  const int N = N_NODES, E = N_EDGES;

  // ---- workspace carve-up (256B aligned) ----
  char* base = (char*)d_ws;
  auto carve = [&](size_t bytes) -> void* {
    void* p = (void*)base;
    base += (bytes + 255) & ~((size_t)255);
    return p;
  };
  _Float16* node16 = (_Float16*)carve((size_t)N * 128 * 2);
  _Float16* msgWT  = (_Float16*)carve((size_t)128 * 192 * 2);
  _Float16* W1T    = (_Float16*)carve((size_t)128 * 768 * 2);
  _Float16* W2T    = (_Float16*)carve((size_t)128 * 384 * 2);
  float*    h_neigh= (float*)   carve((size_t)N * 128 * 4);   // later reused as final h
  float*    deg    = (float*)   carve((size_t)N * 4);
  float*    norm   = (float*)   carve((size_t)N * 4);
  _Float16* x016   = (_Float16*)carve((size_t)N * 256 * 2);
  _Float16* h1_16  = (_Float16*)carve((size_t)N * 256 * 2);
  _Float16* h2_16  = (_Float16*)carve((size_t)N * 256 * 2);
  float*    tmpA   = (float*)   carve((size_t)N * 256 * 4);   // scaled source values
  float*    tmpB   = (float*)   carve((size_t)N * 256 * 4);   // scatter accumulator
  float*    h1f    = (float*)   carve((size_t)N * 128 * 4);   // layer1 output f32
  _Float16* h1h    = (_Float16*)carve((size_t)N * 128 * 2);   // layer1 output f16
  _Float16* h1b16  = (_Float16*)carve((size_t)N * 128 * 2);
  _Float16* h2b16  = (_Float16*)carve((size_t)N * 128 * 2);
  float*    hg     = (float*)   carve(128 * 4);
  float*    hfin   = h_neigh;                                  // reuse (dead after x0)

  const int T = 256;

  // ---- prep: conversions / transposes ----
  k_f32_to_f16<<<cdiv((long long)N * 128, T), T, 0, stream>>>(node_fea, node16, N * 128);
  k_transpose_w<<<cdiv(192 * 128, T), T, 0, stream>>>(msg_W, msgWT, 192, 128);
  k_transpose_w<<<cdiv(768 * 128, T), T, 0, stream>>>(W1, W1T, 768, 128);
  k_transpose_w<<<cdiv(384 * 128, T), T, 0, stream>>>(W2, W2T, 384, 128);

  // ---- edge messages -> h_neigh ----
  k_zero_f32<<<cdiv((long long)N * 128, T), T, 0, stream>>>(h_neigh, N * 128);
  edge_msg_kernel<<<E / (16 * TILES_PER_BLOCK), T, 0, stream>>>(
      node16, edge_fea, src, dst, msgWT, msg_b, h_neigh);

  // ---- degree / norm ----
  k_zero_f32<<<cdiv(N, T), T, 0, stream>>>(deg, N);
  k_deg<<<cdiv(E, T), T, 0, stream>>>(dst, deg, E);
  k_norm<<<cdiv(N, T), T, 0, stream>>>(deg, norm, N);

  // ---- TAGConv layer 1 (D = 256) ----
  k_build_x0<<<cdiv((long long)N * 256, T), T, 0, stream>>>(node_fea, h_neigh, norm, x016, tmpA, N);
  // hop 1
  k_zero_f32<<<cdiv((long long)N * 256, T), T, 0, stream>>>(tmpB, N * 256);
  k_scatter<<<cdiv((long long)E * 64, T), T, 0, stream>>>(tmpA, src, dst, tmpB, E, 6);
  k_postprop<<<cdiv((long long)N * 256, T), T, 0, stream>>>(tmpB, norm, h1_16, tmpA, N, 8);
  // hop 2
  k_zero_f32<<<cdiv((long long)N * 256, T), T, 0, stream>>>(tmpB, N * 256);
  k_scatter<<<cdiv((long long)E * 64, T), T, 0, stream>>>(tmpA, src, dst, tmpB, E, 6);
  k_postprop<<<cdiv((long long)N * 256, T), T, 0, stream>>>(tmpB, norm, h2_16, nullptr, N, 8);
  // GEMM: [N,768] @ W1 + b1, relu
  node_gemm_kernel<<<N / 16, T, 0, stream>>>(x016, h1_16, h2_16, 256, 8, 24, W1T, b1, h1f, h1h);

  // ---- TAGConv layer 2 (D = 128) ----
  k_scale<<<cdiv((long long)N * 128, T), T, 0, stream>>>(h1f, norm, tmpA, N, 7);
  k_zero_f32<<<cdiv((long long)N * 128, T), T, 0, stream>>>(tmpB, N * 128);
  k_scatter<<<cdiv((long long)E * 32, T), T, 0, stream>>>(tmpA, src, dst, tmpB, E, 5);
  k_postprop<<<cdiv((long long)N * 128, T), T, 0, stream>>>(tmpB, norm, h1b16, tmpA, N, 7);
  k_zero_f32<<<cdiv((long long)N * 128, T), T, 0, stream>>>(tmpB, N * 128);
  k_scatter<<<cdiv((long long)E * 32, T), T, 0, stream>>>(tmpA, src, dst, tmpB, E, 5);
  k_postprop<<<cdiv((long long)N * 128, T), T, 0, stream>>>(tmpB, norm, h2b16, nullptr, N, 7);
  // GEMM: [N,384] @ W2 + b2, relu
  node_gemm_kernel<<<N / 16, T, 0, stream>>>(h1h, h1b16, h2b16, 128, 7, 12, W2T, b2, hfin, nullptr);

  // ---- mean pool + predict ----
  k_zero_f32<<<1, 128, 0, stream>>>(hg, 128);
  k_mean_partial<<<256, 128, 0, stream>>>(hfin, hg, N);
  k_predict<<<1, 128, 0, stream>>>(hg, Wp, bp, out, N);
}
